// MultiHeadedAttention_764504178873
// MI455X (gfx1250) — compile-verified, matrix operand order checked by reading, not run on hardware
//
#include <hip/hip_runtime.h>
#include <hip/hip_bf16.h>

typedef __attribute__((ext_vector_type(16))) _Float16 v16h;
typedef __attribute__((ext_vector_type(8)))  _Float16 v8h;
typedef __attribute__((ext_vector_type(8)))  float    v8f;
typedef __attribute__((ext_vector_type(4)))  float    v4f;

#define B_SZ   8
#define S_LEN  1024
#define NHEAD  16
#define DK     64
#define DMODEL 1024
#define WDIM   768
#define MROWS  (B_SZ * S_LEN)   // 8192

// ---------------------------------------------------------------------------
// WMMA wrapper: D = A(16x32 f16) * B(32x16 f16) + C(16x16 f32)
// ---------------------------------------------------------------------------
__device__ __forceinline__ v8f wmma_f16(v16h a, v16h b, v8f c) {
  return __builtin_amdgcn_wmma_f32_16x16x32_f16(
      /*neg_a=*/false, a, /*neg_b=*/false, b,
      /*c_mod=*/(short)0, c, /*reuse_a=*/false, /*reuse_b=*/false);
}

// Fragment loaders. ISA 16-bit A-matrix 16x32 layout (wave32):
//   lane l: row = l&15 ; K-halves [0..7] at kbase=(l>>4)*8, [8..15] at kbase+16
// B fragment of a KxN operand is the same layout applied to the operand's
// transpose, so we always feed these loaders a row-major matrix whose rows
// are the 16 "M/N" lines and whose columns are the 32 K values.
__device__ __forceinline__ v16h load_frag_f16(const _Float16* base, int stride, int lane) {
  int r  = lane & 15;
  int kb = (lane >> 4) * 8;
  const _Float16* p = base + (size_t)r * stride + kb;
  v8h lo = *(const v8h*)(p);
  v8h hi = *(const v8h*)(p + 16);
  v16h f;
#pragma unroll
  for (int i = 0; i < 8; ++i) { f[i] = lo[i]; f[i + 8] = hi[i]; }
  return f;
}

__device__ __forceinline__ v16h load_frag_f32(const float* base, int stride, int lane) {
  int r  = lane & 15;
  int kb = (lane >> 4) * 8;
  const float* p = base + (size_t)r * stride + kb;
  v4f a0 = *(const v4f*)(p);
  v4f a1 = *(const v4f*)(p + 4);
  v4f b0 = *(const v4f*)(p + 16);
  v4f b1 = *(const v4f*)(p + 20);
  v16h f;
#pragma unroll
  for (int i = 0; i < 4; ++i) {
    f[i]      = (_Float16)a0[i];
    f[i + 4]  = (_Float16)a1[i];
    f[i + 8]  = (_Float16)b0[i];
    f[i + 12] = (_Float16)b1[i];
  }
  return f;
}

__device__ __forceinline__ v16h load_frag_lds_f32(const float* base, int stride, int lane) {
  int r  = lane & 15;
  int kb = (lane >> 4) * 8;
  const float* p = base + (size_t)r * stride + kb;
  v16h f;
#pragma unroll
  for (int i = 0; i < 8; ++i) {
    f[i]     = (_Float16)p[i];
    f[i + 8] = (_Float16)p[16 + i];
  }
  return f;
}

// ---------------------------------------------------------------------------
// Weight transpose + f32->f16: W[K x N] row-major -> Wt[N x K] f16
// ---------------------------------------------------------------------------
__global__ void wt_kernel(const float* __restrict__ W, _Float16* __restrict__ Wt,
                          int K, int N) {
  size_t idx = (size_t)blockIdx.x * blockDim.x + threadIdx.x;
  if (idx < (size_t)K * N) {
    int kk = (int)(idx / N);
    int n  = (int)(idx % N);
    Wt[(size_t)n * K + kk] = (_Float16)W[idx];
  }
}

// ---------------------------------------------------------------------------
// QKV projection: dst = split_heads(X[8192x768] @ W + b), f16 outputs.
// One wave -> 16x64 output tile; blockIdx.y == head (N-group of 64).
// mode 0/1: dst[b,h,s,d]   mode 2: dst[b,h,d,s] (V stored transposed)
// ---------------------------------------------------------------------------
__global__ void qkv_gemm(const float* __restrict__ X, const _Float16* __restrict__ Wt,
                         const float* __restrict__ bias, _Float16* __restrict__ dst,
                         int mode) {
  int lane = threadIdx.x & 31;
  int wave = threadIdx.x >> 5;
  int m0   = (blockIdx.x * 4 + wave) * 16;   // row tile in [0, 8192)
  int g    = blockIdx.y;                     // head index / N-group of 64

  v8f acc[4] = {{0.f}, {0.f}, {0.f}, {0.f}};
  for (int kc = 0; kc < WDIM; kc += 32) {
    v16h a = load_frag_f32(X + (size_t)m0 * WDIM + kc, WDIM, lane);
#pragma unroll
    for (int ni = 0; ni < 4; ++ni) {
      v16h bw = load_frag_f16(Wt + (size_t)(g * 64 + ni * 16) * WDIM + kc, WDIM, lane);
      acc[ni] = wmma_f16(a, bw, acc[ni]);
    }
  }

  int nl = lane & 15;
  int mb = (lane >> 4) * 8;
#pragma unroll
  for (int ni = 0; ni < 4; ++ni) {
    float bn = bias[g * 64 + ni * 16 + nl];
#pragma unroll
    for (int r = 0; r < 8; ++r) {
      int m = m0 + mb + r;
      int bb = m >> 10;            // batch
      int ss = m & 1023;           // sequence position
      int d  = ni * 16 + nl;       // within-head dim
      float vv = acc[ni][r] + bn;
      size_t off;
      if (mode == 2) off = ((size_t)(bb * NHEAD + g) * DK + d) * S_LEN + ss;
      else           off = ((size_t)(bb * NHEAD + g) * S_LEN + ss) * DK + d;
      dst[off] = (_Float16)vv;
    }
  }
}

// ---------------------------------------------------------------------------
// Attention: per wave one (b,h,16-row Q tile).
//   scores = Q K^T / 8  -> LDS [16x1024] f32
//   softmax (exp kept unnormalized in LDS; p_attn written normalized f32)
//   ctx = (E V) * (1/rowsum)  -> x[b,s, h*64+d] f16
// 4 waves/block, dynamic LDS = 4 * 16 * 1024 * 4 = 256 KB.
// ---------------------------------------------------------------------------
__global__ void attn_kernel(const _Float16* __restrict__ q, const _Float16* __restrict__ k,
                            const _Float16* __restrict__ vT, float* __restrict__ p_attn,
                            _Float16* __restrict__ x) {
  extern __shared__ float smem[];
  int lane = threadIdx.x & 31;
  int wave = threadIdx.x >> 5;
  float* sE = smem + (size_t)wave * 16 * S_LEN;   // this wave's 16x1024 tile

  int h  = blockIdx.y;
  int b  = blockIdx.z;
  int s0 = (blockIdx.x * 4 + wave) * 16;          // query tile base

  const _Float16* qbh = q  + (size_t)(b * NHEAD + h) * S_LEN * DK;
  const _Float16* kbh = k  + (size_t)(b * NHEAD + h) * S_LEN * DK;
  const _Float16* vbh = vT + (size_t)(b * NHEAD + h) * DK * S_LEN;

  // ---- scores = Q K^T * 0.125 ----
  v16h aQ0 = load_frag_f16(qbh + (size_t)s0 * DK,      DK, lane);  // dk 0..31
  v16h aQ1 = load_frag_f16(qbh + (size_t)s0 * DK + 32, DK, lane);  // dk 32..63
  int nl = lane & 15;
  int mb = (lane >> 4) * 8;
  for (int j = 0; j < S_LEN / 16; ++j) {
    v16h bK0 = load_frag_f16(kbh + (size_t)(j * 16) * DK,      DK, lane);
    v16h bK1 = load_frag_f16(kbh + (size_t)(j * 16) * DK + 32, DK, lane);
    v8f acc = {0.f};
    acc = wmma_f16(aQ0, bK0, acc);
    acc = wmma_f16(aQ1, bK1, acc);
    int col = j * 16 + nl;
#pragma unroll
    for (int r = 0; r < 8; ++r) sE[(size_t)(mb + r) * S_LEN + col] = acc[r] * 0.125f;
  }
  __syncthreads();

  // ---- softmax over each of the 16 rows; 2 lanes per row (512 cols each) ----
  int row = lane & 15;
  int ch  = (lane >> 4) * 512;
  float* sR = sE + (size_t)row * S_LEN + ch;

  float pmax = -3.402823e38f;
  for (int c = 0; c < 512; c += 4) {
    v4f v = *(const v4f*)(sR + c);
    pmax = fmaxf(pmax, fmaxf(fmaxf(v[0], v[1]), fmaxf(v[2], v[3])));
  }
  pmax = fmaxf(pmax, __shfl_xor(pmax, 16, 32));

  float psum = 0.f;
  for (int c = 0; c < 512; c += 4) {
    v4f v = *(const v4f*)(sR + c);
    v4f e;
    e[0] = __expf(v[0] - pmax); e[1] = __expf(v[1] - pmax);
    e[2] = __expf(v[2] - pmax); e[3] = __expf(v[3] - pmax);
    *(v4f*)(sR + c) = e;
    psum += e[0] + e[1] + e[2] + e[3];
  }
  psum += __shfl_xor(psum, 16, 32);
  float inv = 1.0f / psum;

  // normalized probabilities -> p_attn output (f32)
  float* pout = p_attn + ((size_t)(b * NHEAD + h) * S_LEN + (s0 + row)) * S_LEN + ch;
  for (int c = 0; c < 512; c += 4) {
    v4f e = *(const v4f*)(sR + c);
    v4f pv;
    pv[0] = e[0] * inv; pv[1] = e[1] * inv; pv[2] = e[2] * inv; pv[3] = e[3] * inv;
    *(v4f*)(pout + c) = pv;
  }
  __syncthreads();

  // ---- ctx = (E V) * inv ----
  v8f acc[4] = {{0.f}, {0.f}, {0.f}, {0.f}};
  for (int j = 0; j < S_LEN / 32; ++j) {
    v16h a = load_frag_lds_f32(sE + j * 32, S_LEN, lane);
#pragma unroll
    for (int ni = 0; ni < 4; ++ni) {
      v16h bv = load_frag_f16(vbh + (size_t)(ni * 16) * S_LEN + j * 32, S_LEN, lane);
      acc[ni] = wmma_f16(a, bv, acc[ni]);
    }
  }

  _Float16* xb = x + (size_t)b * S_LEN * DMODEL + (size_t)h * DK;
#pragma unroll
  for (int r = 0; r < 8; ++r) {
    int m = mb + r;
    float invm = __shfl(inv, m, 32);   // lane m holds row m's 1/sum
#pragma unroll
    for (int ni = 0; ni < 4; ++ni)
      xb[(size_t)(s0 + m) * DMODEL + ni * 16 + nl] = (_Float16)(acc[ni][r] * invm);
  }
}

// ---------------------------------------------------------------------------
// Output projection: out = x[8192x1024](f16) @ Wo + bo  (f32 out)
// ---------------------------------------------------------------------------
__global__ void out_gemm(const _Float16* __restrict__ X, const _Float16* __restrict__ WoT,
                         const float* __restrict__ bias, float* __restrict__ out) {
  int lane = threadIdx.x & 31;
  int wave = threadIdx.x >> 5;
  int m0   = (blockIdx.x * 4 + wave) * 16;
  int g    = blockIdx.y;

  v8f acc[4] = {{0.f}, {0.f}, {0.f}, {0.f}};
  for (int kc = 0; kc < DMODEL; kc += 32) {
    v16h a = load_frag_f16(X + (size_t)m0 * DMODEL + kc, DMODEL, lane);
#pragma unroll
    for (int ni = 0; ni < 4; ++ni) {
      v16h bw = load_frag_f16(WoT + (size_t)(g * 64 + ni * 16) * DMODEL + kc, DMODEL, lane);
      acc[ni] = wmma_f16(a, bw, acc[ni]);
    }
  }

  int nl = lane & 15;
  int mb = (lane >> 4) * 8;
#pragma unroll
  for (int ni = 0; ni < 4; ++ni) {
    float bn = bias[g * 64 + ni * 16 + nl];
#pragma unroll
    for (int r = 0; r < 8; ++r) {
      int m = m0 + mb + r;
      out[(size_t)m * DMODEL + g * 64 + ni * 16 + nl] = acc[ni][r] + bn;
    }
  }
}

// ---------------------------------------------------------------------------
extern "C" void kernel_launch(void* const* d_in, const int* in_sizes, int n_in,
                              void* d_out, int out_size, void* d_ws, size_t ws_size,
                              hipStream_t stream) {
  (void)in_sizes; (void)n_in; (void)out_size; (void)ws_size;

  const float* Xq = (const float*)d_in[0];
  const float* Xk = (const float*)d_in[1];
  const float* Xv = (const float*)d_in[2];
  const float* Wq = (const float*)d_in[3];
  const float* bq = (const float*)d_in[4];
  const float* Wk = (const float*)d_in[5];
  const float* bk = (const float*)d_in[6];
  const float* Wv = (const float*)d_in[7];
  const float* bv = (const float*)d_in[8];
  const float* Wo = (const float*)d_in[9];
  const float* bo = (const float*)d_in[10];

  float* out_f32 = (float*)d_out;                                   // [8,1024,1024]
  float* p_attn  = out_f32 + (size_t)B_SZ * S_LEN * DMODEL;         // [8,16,1024,1024]

  // workspace carve-up
  char* ws = (char*)d_ws;
  size_t o = 0;
  _Float16* WqT = (_Float16*)(ws + o); o += (size_t)DMODEL * WDIM * 2;
  _Float16* WkT = (_Float16*)(ws + o); o += (size_t)DMODEL * WDIM * 2;
  _Float16* WvT = (_Float16*)(ws + o); o += (size_t)DMODEL * WDIM * 2;
  _Float16* WoT = (_Float16*)(ws + o); o += (size_t)DMODEL * DMODEL * 2;
  _Float16* qh  = (_Float16*)(ws + o); o += (size_t)MROWS * DMODEL * 2;  // [b,h,s,d]
  _Float16* kh  = (_Float16*)(ws + o); o += (size_t)MROWS * DMODEL * 2;  // [b,h,s,d]
  _Float16* vhT = (_Float16*)(ws + o); o += (size_t)MROWS * DMODEL * 2;  // [b,h,d,s]
  _Float16* xc  = (_Float16*)(ws + o); o += (size_t)MROWS * DMODEL * 2;  // [b,s,1024]

  // 1) weight transpose + downconvert
  {
    int n768 = ((WDIM * DMODEL) + 255) / 256;
    wt_kernel<<<n768, 256, 0, stream>>>(Wq, WqT, WDIM, DMODEL);
    wt_kernel<<<n768, 256, 0, stream>>>(Wk, WkT, WDIM, DMODEL);
    wt_kernel<<<n768, 256, 0, stream>>>(Wv, WvT, WDIM, DMODEL);
    int n1024 = ((DMODEL * DMODEL) + 255) / 256;
    wt_kernel<<<n1024, 256, 0, stream>>>(Wo, WoT, DMODEL, DMODEL);
  }

  // 2) QKV projections (head-split layouts)
  dim3 ggrid(MROWS / 64, NHEAD);
  qkv_gemm<<<ggrid, 128, 0, stream>>>(Xq, WqT, bq, qh, 0);
  qkv_gemm<<<ggrid, 128, 0, stream>>>(Xk, WkT, bk, kh, 1);
  qkv_gemm<<<ggrid, 128, 0, stream>>>(Xv, WvT, bv, vhT, 2);

  // 3) attention (+p_attn output)
  dim3 agrid(S_LEN / 16 / 4, NHEAD, B_SZ);
  size_t lds = (size_t)4 * 16 * S_LEN * sizeof(float);   // 256 KB
  attn_kernel<<<agrid, 128, lds, stream>>>(qh, kh, vhT, p_attn, xc);

  // 4) output projection
  out_gemm<<<ggrid, 128, 0, stream>>>(xc, WoT, bo, out_f32);
}
